// MoEBiEncoder_17772574671523
// MI455X (gfx1250) — matrix operand order repaired and checked by reading, not compiled, not run on hardware
//
#include <hip/hip_runtime.h>

typedef __attribute__((ext_vector_type(16))) __bf16 bf16x16;
typedef __attribute__((ext_vector_type(8)))  __bf16 bf16x8;
typedef __attribute__((ext_vector_type(8)))  float  f32x8;

#define BH   8192
#define HDIM 768
#define HH   384
#define NE   8

#define BM 128
#define BN 64
#define BK 32
#define LDA 56   // BK + 24 pad: row stride = 112B (16B aligned, conflict-light)
#define LDB 56

__device__ __forceinline__ __bf16 f2bf(float f) {
  unsigned u = __builtin_bit_cast(unsigned, f);
  unsigned r = u + 0x7FFFu + ((u >> 16) & 1u);   // round-to-nearest-even
  unsigned short s = (unsigned short)(r >> 16);
  return __builtin_bit_cast(__bf16, s);
}

__global__ void cvt_f32_bf16(const float* __restrict__ in, __bf16* __restrict__ out, int n) {
  int stride = gridDim.x * blockDim.x;
  for (int i = blockIdx.x * blockDim.x + threadIdx.x; i < n; i += stride)
    out[i] = f2bf(in[i]);
}

__global__ void zero_counts(int* counts) {
  if (threadIdx.x < NE) counts[threadIdx.x] = 0;
}

union AFrag { bf16x16 v; bf16x8 h[2]; };

// MODE 0: x1 = relu(A@B + bias) -> fp32, rows implicit
// MODE 1: h  = relu(Arows@B[e] + bias[e]) -> bf16, rows gathered via rowlist
// MODE 2: pre = gate[b] * (Arows@B[e] + bias[e]) -> fp32, rows gathered
template <int MODE>
__global__ __launch_bounds__(256)
void gemm_wmma(const __bf16* __restrict__ A,
               const __bf16* __restrict__ Bw,
               const float*  __restrict__ bias,
               float*        __restrict__ outF,
               __bf16*       __restrict__ outBF,
               const int*    __restrict__ rowlist,
               const int*    __restrict__ counts,
               const float*  __restrict__ gate,
               int K, int Ntot) {
  __shared__ __align__(16) __bf16 As[BM * LDA];
  __shared__ __align__(16) __bf16 Bs[BN * LDB];   // transposed: [n][k]

  const int tid  = threadIdx.x;
  const int lane = tid & 31;
  const int w    = tid >> 5;
  const int wm   = w & 3;      // 4 waves along M (32 rows each)
  const int wn   = w >> 2;     // 2 waves along N (32 cols each)
  const int mtile = blockIdx.x;
  const int n0    = blockIdx.y * BN;
  const int e     = (MODE == 0) ? 0 : blockIdx.z;

  int m_count = BM;
  const int* rl = nullptr;
  if constexpr (MODE != 0) {
    int cnt = counts[e];
    if (mtile * BM >= cnt) return;
    m_count = min(BM, cnt - mtile * BM);
    rl = rowlist + e * BH + mtile * BM;
  }

  const __bf16* Bbase = Bw + (size_t)e * K * Ntot;
  const float*  bbase = bias + (size_t)e * Ntot;

  // Global->LDS load coordinates: A 128x32 (16 bf16/thread), B 32x64 (8/thread)
  const int ai = tid >> 1;
  const int ak = (tid & 1) * 16;
  int arow;
  if constexpr (MODE == 0) arow = mtile * BM + ai;
  else                     arow = (ai < m_count) ? rl[ai] : rl[0];
  const __bf16* aptr = A + (size_t)arow * K + ak;
  const unsigned lds_a = (unsigned)(uintptr_t)(&As[ai * LDA + ak]);

  const int bk = tid >> 3;
  const int bn = (tid & 7) * 8;
  const __bf16* bptr = Bbase + (size_t)bk * Ntot + n0 + bn;

  f32x8 acc[2][2] = {};

  const int ksteps = K / BK;
  for (int ks = 0; ks < ksteps; ++ks) {
    // A tile: async copy global -> LDS (32B per thread), bypasses VGPRs
    asm volatile("global_load_async_to_lds_b128 %0, %1, off"
                 :: "v"(lds_a), "v"(aptr) : "memory");
    asm volatile("global_load_async_to_lds_b128 %0, %1, off offset:16"
                 :: "v"(lds_a), "v"(aptr) : "memory");

    // B tile: load row-major, scatter-transpose into LDS [n][k]
    bf16x8 bv = *(const bf16x8*)(bptr);
    #pragma unroll
    for (int j = 0; j < 8; ++j)
      Bs[(bn + j) * LDB + bk] = bv[j];

    aptr += BK;
    bptr += (size_t)BK * Ntot;
    if (ks + 1 < ksteps) __builtin_prefetch(bptr, 0, 1);

    asm volatile("s_wait_asynccnt 0x0" ::: "memory");
    __syncthreads();

    // A 16x32 bf16 fragment: lanes 0-15 -> K 0..7 & 16..23; lanes 16-31 -> K 8..15 & 24..31
    AFrag afr[2], bfr[2];
    const int ar  = lane & 15;
    const int akh = (lane >> 4) * 8;
    #pragma unroll
    for (int ms = 0; ms < 2; ++ms) {
      const __bf16* p = &As[(wm * 32 + ms * 16 + ar) * LDA + akh];
      afr[ms].h[0] = *(const bf16x8*)(p);
      afr[ms].h[1] = *(const bf16x8*)(p + 16);
    }
    // B 32x16 fragment: lanes 0-15 -> col=lane, K 0..7 & 16..23; lanes 16-31 -> K 8..15 & 24..31
    const int bc  = lane & 15;
    const int bkh = (lane >> 4) * 8;
    #pragma unroll
    for (int ns = 0; ns < 2; ++ns) {
      const __bf16* p = &Bs[(wn * 32 + ns * 16 + bc) * LDB + bkh];
      bfr[ns].h[0] = *(const bf16x8*)(p);
      bfr[ns].h[1] = *(const bf16x8*)(p + 16);
    }
    #pragma unroll
    for (int ms = 0; ms < 2; ++ms)
      #pragma unroll
      for (int ns = 0; ns < 2; ++ns)
        acc[ms][ns] = __builtin_amdgcn_wmma_f32_16x16x32_bf16(
            false, afr[ms].v, false, bfr[ns].v, (short)0, acc[ms][ns],
            false, false);

    __syncthreads();
  }

  // Epilogue. C layout: VGPR v, lanes 0-15 -> M=v, N=lane; lanes 16-31 -> M=v+8
  const int lr = lane & 15;
  const int lh = lane >> 4;
  #pragma unroll
  for (int ns = 0; ns < 2; ++ns) {
    const int col = n0 + wn * 32 + ns * 16 + lr;
    const float bvv = bbase[col];
    #pragma unroll
    for (int ms = 0; ms < 2; ++ms) {
      #pragma unroll
      for (int v = 0; v < 8; ++v) {
        const int rloc = wm * 32 + ms * 16 + v + lh * 8;
        float val = acc[ms][ns][v] + bvv;
        if constexpr (MODE == 0) {
          outF[(size_t)(mtile * BM + rloc) * HH + col] = fmaxf(val, 0.0f);
        } else if constexpr (MODE == 1) {
          if (rloc < m_count) {
            int b = rl[rloc];
            outBF[(size_t)b * HH + col] = f2bf(fmaxf(val, 0.0f));
          }
        } else {
          if (rloc < m_count) {
            int b = rl[rloc];
            outF[(size_t)b * HDIM + col] = val * gate[b];
          }
        }
      }
    }
  }
}

__global__ __launch_bounds__(256)
void gating_kernel(const float* __restrict__ x1, const float* __restrict__ w3,
                   const float* __restrict__ b3, int* __restrict__ eidx,
                   float* __restrict__ gate, int* __restrict__ counts,
                   int* __restrict__ rowlist) {
  const int wave = threadIdx.x >> 5;
  const int lane = threadIdx.x & 31;
  const int b = blockIdx.x * 8 + wave;
  float acc[NE];
  #pragma unroll
  for (int e = 0; e < NE; ++e) acc[e] = 0.f;
  const float* xr = x1 + (size_t)b * HH;
  for (int f = lane; f < HH; f += 32) {
    float xv = xr[f];
    const float* wr = w3 + f * NE;
    #pragma unroll
    for (int e = 0; e < NE; ++e) acc[e] += xv * wr[e];
  }
  #pragma unroll
  for (int e = 0; e < NE; ++e)
    for (int off = 16; off > 0; off >>= 1)
      acc[e] += __shfl_xor(acc[e], off, 32);
  if (lane == 0) {
    float mx = -1e30f; int am = 0;
    #pragma unroll
    for (int e = 0; e < NE; ++e) {
      float v = acc[e] + b3[e];
      acc[e] = v;
      if (v > mx) { mx = v; am = e; }     // strict '>' keeps first max (argmax tie rule)
    }
    float s = 0.f;
    #pragma unroll
    for (int e = 0; e < NE; ++e) s += __expf(acc[e] - mx);
    eidx[b] = am;
    gate[b] = 1.0f / s;                    // softmax prob of the argmax
    int pos = atomicAdd(&counts[am], 1);
    rowlist[am * BH + pos] = b;
  }
}

__global__ __launch_bounds__(256)
void finalize_kernel(const float* __restrict__ pre, const float* __restrict__ q,
                     float* __restrict__ out) {
  __shared__ float red[256];
  const int b = blockIdx.x;
  const float* pr = pre + (size_t)b * HDIM;
  float s = 0.f;
  for (int c = threadIdx.x; c < HDIM; c += 256) { float v = pr[c]; s += v * v; }
  red[threadIdx.x] = s;
  __syncthreads();
  for (int off = 128; off > 0; off >>= 1) {
    if (threadIdx.x < off) red[threadIdx.x] += red[threadIdx.x + off];
    __syncthreads();
  }
  const float rn = 1.0f / fmaxf(sqrtf(red[0]), 1e-6f);
  for (int c = threadIdx.x; c < HDIM; c += 256)
    out[(size_t)b * HDIM + c] = pr[c] * rn + q[(size_t)b * HDIM + c];
}

extern "C" void kernel_launch(void* const* d_in, const int* in_sizes, int n_in,
                              void* d_out, int out_size, void* d_ws, size_t ws_size,
                              hipStream_t stream) {
  (void)in_sizes; (void)n_in; (void)out_size; (void)ws_size;
  const float* q      = (const float*)d_in[0];
  const float* cls_w1 = (const float*)d_in[1];
  const float* cls_b1 = (const float*)d_in[2];
  const float* cls_w3 = (const float*)d_in[3];
  const float* cls_b3 = (const float*)d_in[4];
  const float* exp_w1 = (const float*)d_in[5];
  const float* exp_b1 = (const float*)d_in[6];
  const float* exp_w2 = (const float*)d_in[7];
  const float* exp_b2 = (const float*)d_in[8];
  float* out = (float*)d_out;

  char* ws = (char*)d_ws;
  size_t off = 0;
  auto take = [&](size_t bytes) -> char* {
    char* p = ws + off;
    off = (off + bytes + 255) & ~(size_t)255;
    return p;
  };
  __bf16* qb     = (__bf16*)take((size_t)BH * HDIM * 2);
  __bf16* w1b    = (__bf16*)take((size_t)HDIM * HH * 2);
  __bf16* ew1b   = (__bf16*)take((size_t)NE * HDIM * HH * 2);
  __bf16* ew2b   = (__bf16*)take((size_t)NE * HH * HDIM * 2);
  float*  x1     = (float*) take((size_t)BH * HH * 4);
  __bf16* hbuf   = (__bf16*)take((size_t)BH * HH * 2);
  float*  opre   = (float*) take((size_t)BH * HDIM * 4);
  int*    eidx   = (int*)   take((size_t)BH * 4);
  float*  gate   = (float*) take((size_t)BH * 4);
  int*    counts = (int*)   take((size_t)NE * 4);
  int*    rlist  = (int*)   take((size_t)NE * BH * 4);

  cvt_f32_bf16<<<2048, 256, 0, stream>>>(q, qb, BH * HDIM);
  cvt_f32_bf16<<<512, 256, 0, stream>>>(cls_w1, w1b, HDIM * HH);
  cvt_f32_bf16<<<2048, 256, 0, stream>>>(exp_w1, ew1b, NE * HDIM * HH);
  cvt_f32_bf16<<<2048, 256, 0, stream>>>(exp_w2, ew2b, NE * HH * HDIM);
  zero_counts<<<1, 32, 0, stream>>>(counts);

  // x1 = relu(q @ cls_w1 + b1)
  gemm_wmma<0><<<dim3(BH / BM, HH / BN, 1), 256, 0, stream>>>(
      qb, w1b, cls_b1, x1, nullptr, nullptr, nullptr, nullptr, HDIM, HH);

  // softmax/argmax gating + per-expert row compaction
  gating_kernel<<<BH / 8, 256, 0, stream>>>(x1, cls_w3, cls_b3, eidx, gate, counts, rlist);

  // h = relu(q_rows @ exp_w1[e] + exp_b1[e])
  gemm_wmma<1><<<dim3(BH / BM, HH / BN, NE), 256, 0, stream>>>(
      qb, ew1b, exp_b1, nullptr, hbuf, rlist, counts, nullptr, HDIM, HH);

  // pre = gate * (h_rows @ exp_w2[e] + exp_b2[e])
  gemm_wmma<2><<<dim3(BH / BM, HDIM / BN, NE), 256, 0, stream>>>(
      hbuf, ew2b, exp_b2, opre, nullptr, rlist, counts, gate, HH, HDIM);

  // out = pre / max(||pre||, 1e-6) + q
  finalize_kernel<<<BH, 256, 0, stream>>>(opre, q, out);
}
